// SimAKT_87737591922947
// MI455X (gfx1250) — compile-verified, matrix operand order checked by reading, not run on hardware
//
#include <hip/hip_runtime.h>

// ---------------------------------------------------------------------------
// SimAKT block on CDNA5 (gfx1250): bf16 WMMA GEMMs with async global->LDS
// A-tile staging (ASYNCcnt path), 4 N-tiles per wave, fused row-wise kernels.
// ---------------------------------------------------------------------------

#define Bq 8
#define Sq 2048
#define Eq 512
#define BSq (Bq * Sq)            // 16384
#define BSEq ((long long)BSq * Eq)

typedef __attribute__((ext_vector_type(16))) __bf16          v16bf;
typedef __attribute__((ext_vector_type(16))) unsigned short  v16u16;
typedef __attribute__((ext_vector_type(8)))  float           v8f;

__device__ __forceinline__ unsigned short f2bf(float f) {
    unsigned u = __builtin_bit_cast(unsigned, f);
    u += 0x7FFFu + ((u >> 16) & 1u);            // round-to-nearest-even
    return (unsigned short)(u >> 16);
}

__device__ __forceinline__ float sigf(float x) {
    return 1.0f / (1.0f + __expf(-x));
}

// ---- CDNA5 async global->LDS copy (16B per lane), tracked by ASYNCcnt ------
__device__ __forceinline__ void async_g2l_b128(unsigned lds_off, const void* gptr) {
    asm volatile("global_load_async_to_lds_b128 %0, %1, off"
                 :: "v"(lds_off), "v"(gptr) : "memory");
}
__device__ __forceinline__ void wait_async0() {
    asm volatile("s_wait_asynccnt 0x0" ::: "memory");
}

// A/B fragment for v_wmma_f32_16x16x32_bf16 from a row-major [16 x 32] tile.
// lanes 0-15  : row = lane,    K = {0..7, 16..23}
// lanes 16-31 : row = lane-16, K = {8..15, 24..31}
__device__ __forceinline__ v16bf frag_bf16(const unsigned short* m, long long ld,
                                           long long row, long long k0, int half) {
    const unsigned short* p = m + row * ld + k0 + (long long)half * 8;
    v16u16 u;
#pragma unroll
    for (int i = 0; i < 8; ++i) u[i] = p[i];
#pragma unroll
    for (int i = 0; i < 8; ++i) u[8 + i] = p[16 + i];
    return __builtin_bit_cast(v16bf, u);
}

// A fragment out of the LDS-staged [16 x 32] tile (bf16 layout).
__device__ __forceinline__ v16bf frag_lds_bf16(const unsigned short* sa, int r, int half) {
    const unsigned short* p = sa + r * 32 + half * 8;
    v16u16 u;
#pragma unroll
    for (int i = 0; i < 8; ++i) u[i] = p[i];
#pragma unroll
    for (int i = 0; i < 8; ++i) u[8 + i] = p[16 + i];
    return __builtin_bit_cast(v16bf, u);
}

// A fragment out of the LDS-staged [16 x 32] tile (fp32 -> bf16 convert).
__device__ __forceinline__ v16bf frag_lds_f32(const float* sf, int r, int half) {
    const float* p = sf + r * 32 + half * 8;
    v16u16 u;
#pragma unroll
    for (int i = 0; i < 8; ++i) u[i] = f2bf(p[i]);
#pragma unroll
    for (int i = 0; i < 8; ++i) u[8 + i] = f2bf(p[16 + i]);
    return __builtin_bit_cast(v16bf, u);
}

// --------------------------- generic WMMA GEMM ------------------------------
// Y[M,N] = X[M,K] @ Wt[N,K]^T (+bias) with fused epilogues.
// ld(A) == ld(Bt) == K for every call site in this pipeline.
// Block = 8 waves; block computes 16 rows x 512 cols; wave owns 4 x (16x16)
// N-tiles; the 16x32 A tile is staged once per K-step via async LDS copy.
enum { EPI_BIAS_F32 = 0, EPI_RELU_BF16 = 1, EPI_VT_BF16 = 2,
       EPI_SIGMIX_F32 = 3, EPI_BF16 = 4 };

template <int EPI, bool AF32>
__global__ __launch_bounds__(256) void gemm16(const void* __restrict__ A,
                                              const unsigned short* __restrict__ Bt,
                                              const float* __restrict__ bias,
                                              void* __restrict__ Y,
                                              long long M, long long N, long long K) {
    __shared__ __align__(16) unsigned char smem[2048];   // 16x32 fp32 worst case

    const int lane = threadIdx.x & 31;
    const int wave = threadIdx.x >> 5;
    const int half = lane >> 4;
    const int r    = lane & 15;

    const long long mTile = (long long)blockIdx.x * 16;
    const long long nBase = (long long)blockIdx.y * 512 + (long long)wave * 64;

    const unsigned lds_base = (unsigned)(size_t)(&smem[0]);

    v8f acc[4];
#pragma unroll
    for (int j = 0; j < 4; ++j)
#pragma unroll
        for (int i = 0; i < 8; ++i) acc[j][i] = 0.0f;

    for (long long kk = 0; kk < K; kk += 32) {
        // ---- stage A tile [16 x 32] into LDS (wave 0, async path) ----
        if (wave == 0) {
            if (AF32) {
                const int row = lane >> 3;           // 0..3
                const int c   = lane & 7;            // 16B chunk within 128B row
                const float* g0 = (const float*)A + (mTile + row) * K + kk + c * 4;
                const unsigned l0 = lds_base + row * 128 + c * 16;
#pragma unroll
                for (int rg = 0; rg < 4; ++rg)
                    async_g2l_b128(l0 + rg * 4 * 128, g0 + (long long)rg * 4 * K);
            } else {
                const int row = lane >> 2;           // 0..7
                const int c   = lane & 3;            // 16B chunk within 64B row
                const unsigned short* g0 =
                    (const unsigned short*)A + (mTile + row) * K + kk + c * 8;
                const unsigned l0 = lds_base + row * 64 + c * 16;
                async_g2l_b128(l0, g0);
                async_g2l_b128(l0 + 8 * 64, g0 + 8 * K);
            }
            wait_async0();
        }
        __syncthreads();                              // LDS tile visible to all waves

        const v16bf a = AF32 ? frag_lds_f32((const float*)smem, r, half)
                             : frag_lds_bf16((const unsigned short*)smem, r, half);

#pragma unroll
        for (int j = 0; j < 4; ++j) {
            const long long brow = nBase + j * 16 + r;
            if (kk + 32 < K)
                __builtin_prefetch(Bt + brow * K + kk + 32, 0, 1);
            const v16bf b = frag_bf16(Bt, K, brow, kk, half);
            acc[j] = __builtin_amdgcn_wmma_f32_16x16x32_bf16(
                         false, a, false, b, (short)0, acc[j], false, false);
        }
        __syncthreads();                              // all reads done before restage
    }

#pragma unroll
    for (int j = 0; j < 4; ++j) {
        const long long n = nBase + j * 16 + r;
        const float bv = (bias != nullptr) ? bias[n] : 0.0f;
#pragma unroll
        for (int t = 0; t < 8; ++t) {
            const long long m = mTile + t + 8 * half;
            const float v = acc[j][t] + bv;
            if (EPI == EPI_BIAS_F32) {
                ((float*)Y)[m * N + n] = v;
            } else if (EPI == EPI_RELU_BF16) {
                ((unsigned short*)Y)[m * N + n] = f2bf(fmaxf(v, 0.0f));
            } else if (EPI == EPI_BF16) {
                ((unsigned short*)Y)[m * N + n] = f2bf(v);
            } else if (EPI == EPI_VT_BF16) {
                // m = b*S + jrow ; store V transposed: Vt[b][n][jrow]
                const long long bidx = m >> 11, jrow = m & (Sq - 1);
                ((unsigned short*)Y)[(bidx * N + n) * Sq + jrow] = f2bf(v);
            } else { // EPI_SIGMIX_F32 : averaged multi-view sigmoid of cosine sim
                const float c = acc[j][t];
                const float a0 = sigf(c);
                const float a1 = sigf(c * c);
                const float a2 = sigf(__expf(c - 1.0f));
                const float a3 = sigf(0.5f * c + 0.5f);
                ((float*)Y)[m * N + n] = 0.25f * (a0 + a1 + a2 + a3);
            }
        }
    }
}

// --------------------------- row-wise helpers -------------------------------

__device__ __forceinline__ float block_sum(float v, float* sm) {
    const int t = threadIdx.x;
    sm[t] = v;
    __syncthreads();
    for (int s = 128; s > 0; s >>= 1) {
        if (t < s) sm[t] += sm[t + s];
        __syncthreads();
    }
    const float r = sm[0];
    __syncthreads();
    return r;
}

// L2-normalize each length-E row of trajectories, emit bf16.
__global__ __launch_bounds__(256) void norm_rows_bf16(const float* __restrict__ traj,
                                                      unsigned short* __restrict__ tn) {
    __shared__ float sm[256];
    const long long row = blockIdx.x;
    const int t = threadIdx.x;
    const float* p = traj + row * Eq;
    const float a0 = p[t], a1 = p[t + 256];
    const float ss = block_sum(a0 * a0 + a1 * a1, sm);
    const float inv = 1.0f / fmaxf(sqrtf(ss), 1e-12f);
    unsigned short* o = tn + row * Eq;
    o[t]       = f2bf(a0 * inv);
    o[t + 256] = f2bf(a1 * inv);
}

__global__ __launch_bounds__(256) void cvt_bf16(const float* __restrict__ in,
                                                unsigned short* __restrict__ out,
                                                long long n) {
    const long long i = (long long)blockIdx.x * 256 + threadIdx.x;
    if (i < n) out[i] = f2bf(in[i]);
}

// Wt[n*E + k] = bf16(W[k*E + n])
__global__ __launch_bounds__(256) void transpose_cvt(const float* __restrict__ W,
                                                     unsigned short* __restrict__ Wt) {
    const int idx = blockIdx.x * 256 + threadIdx.x; // E*E total
    const int n = idx / Eq, k = idx % Eq;
    Wt[idx] = f2bf(W[(long long)k * Eq + n]);
}

// Fused softmax -> strict-upper mask -> softmax for one row of P[S,S].
// Values after the sigmoid mix are in (0,1) so no max-subtraction is needed;
// the second softmax's exponents are in [1,e].  All-masked row (i==S-1) ->
// uniform 1/S, matching jax softmax over a row of -1e9.
__global__ __launch_bounds__(256) void double_softmax(float* __restrict__ P) {
    __shared__ float sm[256];
    const int i = blockIdx.x;
    float* row = P + (long long)i * Sq;
    const int t = threadIdx.x;
    float v[8];
    float s1 = 0.0f;
#pragma unroll
    for (int u = 0; u < 8; ++u) {
        v[u] = __expf(row[t + 256 * u]);
        s1 += v[u];
    }
    s1 = block_sum(s1, sm);
    const float inv1 = 1.0f / s1;
    float s2 = 0.0f;
#pragma unroll
    for (int u = 0; u < 8; ++u) {
        const int j = t + 256 * u;
        const float p1 = v[u] * inv1;
        const float w = (j > i) ? __expf(p1) : 0.0f;
        v[u] = w;
        s2 += w;
    }
    s2 = block_sum(s2, sm);
    if (s2 > 0.0f) {
        const float inv2 = 1.0f / s2;
#pragma unroll
        for (int u = 0; u < 8; ++u) row[t + 256 * u] = v[u] * inv2;
    } else {
        const float uni = 1.0f / (float)Sq;
#pragma unroll
        for (int u = 0; u < 8; ++u) row[t + 256 * u] = uni;
    }
}

// x = LayerNorm(base + delta) * g + beta ; optionally also emit bf16(x).
__global__ __launch_bounds__(256) void add_ln(const float* __restrict__ base,
                                              const float* __restrict__ delta,
                                              const float* __restrict__ g,
                                              const float* __restrict__ beta,
                                              float* __restrict__ xout,
                                              unsigned short* __restrict__ xbf) {
    __shared__ float sm[256];
    const long long row = blockIdx.x;
    const int t = threadIdx.x;
    const float* b0 = base + row * Eq;
    const float* d0 = delta + row * Eq;
    const float v0 = b0[t] + d0[t];
    const float v1 = b0[t + 256] + d0[t + 256];
    const float mu = block_sum(v0 + v1, sm) * (1.0f / Eq);
    const float c0 = v0 - mu, c1 = v1 - mu;
    const float var = block_sum(c0 * c0 + c1 * c1, sm) * (1.0f / Eq);
    const float inv = rsqrtf(var + 1e-5f);
    const float y0 = c0 * inv * g[t] + beta[t];
    const float y1 = c1 * inv * g[t + 256] + beta[t + 256];
    xout[row * Eq + t]       = y0;
    xout[row * Eq + t + 256] = y1;
    if (xbf != nullptr) {
        xbf[row * Eq + t]       = f2bf(y0);
        xbf[row * Eq + t + 256] = f2bf(y1);
    }
}

// ---------------------------------------------------------------------------

extern "C" void kernel_launch(void* const* d_in, const int* in_sizes, int n_in,
                              void* d_out, int out_size, void* d_ws, size_t ws_size,
                              hipStream_t stream) {
    (void)in_sizes; (void)n_in; (void)out_size; (void)ws_size;

    const float* q     = (const float*)d_in[0];
    const float* k     = (const float*)d_in[1];
    const float* traj  = (const float*)d_in[2];
    const float* vW    = (const float*)d_in[3];
    const float* vb    = (const float*)d_in[4];
    const float* outW  = (const float*)d_in[5];
    const float* outb  = (const float*)d_in[6];
    const float* ln1g  = (const float*)d_in[7];
    const float* ln1b  = (const float*)d_in[8];
    const float* w1    = (const float*)d_in[9];
    const float* b1    = (const float*)d_in[10];
    const float* w2    = (const float*)d_in[11];
    const float* b2    = (const float*)d_in[12];
    const float* ln2g  = (const float*)d_in[13];
    const float* ln2b  = (const float*)d_in[14];

    char* w = (char*)d_ws;
    const size_t MB = 1u << 20;
    unsigned short* tn_bf = (unsigned short*)(w);               // 16 MB
    unsigned short* k_bf  = (unsigned short*)(w + 16 * MB);     // 16 MB
    float*          Pbuf  = (float*)(w + 16 * MB);              // aliases k_bf (k_bf dead after V proj)
    unsigned short* vWt   = (unsigned short*)(w + 32 * MB);     // 4 x 0.5 MB
    unsigned short* outWt = vWt   + (size_t)Eq * Eq;
    unsigned short* w1t   = outWt + (size_t)Eq * Eq;
    unsigned short* w2t   = w1t   + (size_t)Eq * Eq;
    unsigned short* Vt    = (unsigned short*)(w + 34 * MB);     // 16 MB  [B][E][S]
    unsigned short* attb  = (unsigned short*)(w + 50 * MB);     // 16 MB  [B][S][E]
    float*          ybuf  = (float*)(w + 66 * MB);              // 32 MB
    float*          xbuf  = (float*)(w + 98 * MB);              // 32 MB
    unsigned short* x_bf  = (unsigned short*)(w + 130 * MB);    // 16 MB
    unsigned short* h_bf  = (unsigned short*)(w + 146 * MB);    // 16 MB -> 162 MB total

    // 1) operand preparation
    norm_rows_bf16<<<BSq, 256, 0, stream>>>(traj, tn_bf);
    cvt_bf16<<<(unsigned)((BSEq + 255) / 256), 256, 0, stream>>>(k, k_bf, BSEq);
    transpose_cvt<<<Eq * Eq / 256, 256, 0, stream>>>(vW,   vWt);
    transpose_cvt<<<Eq * Eq / 256, 256, 0, stream>>>(outW, outWt);
    transpose_cvt<<<Eq * Eq / 256, 256, 0, stream>>>(w1,   w1t);
    transpose_cvt<<<Eq * Eq / 256, 256, 0, stream>>>(w2,   w2t);

    // 2) V = k @ v_W + v_b, stored transposed per batch: Vt[b][e][j]
    gemm16<EPI_VT_BF16, false><<<dim3(BSq / 16, 1), 256, 0, stream>>>(
        k_bf, vWt, vb, Vt, BSq, Eq, Eq);

    // 3) attention per batch (stream-serialized; P buffer reused)
    for (int b = 0; b < Bq; ++b) {
        const long long off = (long long)b * Sq * Eq;
        gemm16<EPI_SIGMIX_F32, false><<<dim3(Sq / 16, Sq / 512), 256, 0, stream>>>(
            tn_bf + off, tn_bf + off, nullptr, Pbuf, Sq, Sq, Eq);
        double_softmax<<<Sq, 256, 0, stream>>>(Pbuf);
        gemm16<EPI_BF16, true><<<dim3(Sq / 16, 1), 256, 0, stream>>>(
            Pbuf, Vt + (long long)b * Eq * Sq, nullptr, attb + off, Sq, Eq, Sq);
    }

    // 4) output projection + residual LN
    gemm16<EPI_BIAS_F32, false><<<dim3(BSq / 16, 1), 256, 0, stream>>>(
        attb, outWt, outb, ybuf, BSq, Eq, Eq);
    add_ln<<<BSq, 256, 0, stream>>>(q, ybuf, ln1g, ln1b, xbuf, x_bf);

    // 5) FFN + residual LN -> d_out
    gemm16<EPI_RELU_BF16, false><<<dim3(BSq / 16, 1), 256, 0, stream>>>(
        x_bf, w1t, b1, h_bf, BSq, Eq, Eq);
    gemm16<EPI_BIAS_F32, false><<<dim3(BSq / 16, 1), 256, 0, stream>>>(
        h_bf, w2t, b2, ybuf, BSq, Eq, Eq);
    add_ln<<<BSq, 256, 0, stream>>>(xbuf, ybuf, ln2g, ln2b, (float*)d_out, nullptr);
}
